// PluckerBigramAttention_60739427500738
// MI455X (gfx1250) — matrix operand order repaired and checked by reading, not compiled, not run on hardware
//
#include <hip/hip_runtime.h>
#include <hip/hip_bf16.h>
#include <math.h>
#include <stdint.h>

// ===========================================================================
// PluckerBigramAttention on MI455X (gfx1250, wave32, WMMA + TDM).
//
//   1) cvt x -> f16, build x_prev (zero row 0) -> f16
//   2) pack all weights into WMMA B-fragment layout (f16)
//   3) qk kernel: WMMA GEMMs (f16) -> LDS -> Plucker normalize -> q8 / Jk8
//   4) v kernel: WMMA GEMM, stores V transposed (d-major) in f16
//   5) flash attention: logits^T via v_wmma_f32_16x16x4_f32 (fp32, K=4 x2
//      covers the padded 6-dim Plucker product), online softmax in-lane,
//      V^T tiles DMA'd to LDS by the Tensor Data Mover (double buffered,
//      s_wait_tensorcnt pipelining), P@V via v_wmma_f32_16x16x32_f16 on
//      o^T = V^T P^T
//   6) out GEMM: WMMA f16, fp32 stores
//
// Workspace requirement: ~43 MB.
// ===========================================================================

typedef __attribute__((ext_vector_type(16))) _Float16 v16h;
typedef __attribute__((ext_vector_type(8)))  _Float16 v8h;
typedef __attribute__((ext_vector_type(8)))  float    v8f;
typedef __attribute__((ext_vector_type(2)))  float    v2f;
typedef __attribute__((ext_vector_type(4)))  unsigned int u32x4;
typedef __attribute__((ext_vector_type(8)))  int      i32x8;
typedef __attribute__((ext_vector_type(4)))  int      i32x4;

namespace pba {
constexpr int B = 2, T = 2048, D = 1024, H = 16;
constexpr int TB = B * T;                       // 4096 rows total
constexpr float SCALE = 0.408248290463863f;     // 6^-0.5
}

#if __has_builtin(__builtin_amdgcn_tensor_load_to_lds) && \
    __has_builtin(__builtin_amdgcn_s_wait_tensorcnt)
#define PBA_HAVE_TDM 1
#else
#define PBA_HAVE_TDM 0
#endif

#define DEV static __device__ __forceinline__

// A fragment (16x32 f16): lane L holds row M=L%16; element i -> K=kbase+i+16*(L>=16).
// Two 16B loads (global_load_b128 / ds_load_b128).
DEV v16h load_a_f16(const _Float16* mat, int ld, int row, int kbase, int lane) {
  const _Float16* p = mat + (size_t)row * ld + kbase + ((lane & 16) ? 16 : 0);
  union { v16h v; v8h h[2]; } u;
  u.h[0] = *(const v8h*)p;
  u.h[1] = *(const v8h*)(p + 8);
  return u.v;
}

// B fragment from pre-packed weights: one contiguous 32B read per lane.
DEV v16h load_b_frag(const _Float16* packed, int kb, int nb, int NB, int lane) {
  return *(const v16h*)(packed + ((((size_t)kb * NB + nb) * 32) + lane) * 16);
}

DEV void zero_acc4(v8f acc[4]) {
  v8f z = {0.f, 0.f, 0.f, 0.f, 0.f, 0.f, 0.f, 0.f};
  acc[0] = z; acc[1] = z; acc[2] = z; acc[3] = z;
}

// 16 rows x 64 cols GEMM piece: A row-major f16 (ld=ldA), B pre-packed.
DEV void gemm_16x64(const _Float16* A, int ldA, int row0,
                    const _Float16* pB, int nbBase, int NB, int nKb,
                    int lane, v8f acc[4]) {
  for (int kb = 0; kb < nKb; ++kb) {
    v16h a = load_a_f16(A, ldA, row0 + (lane & 15), kb * 32, lane);
#pragma unroll
    for (int nb = 0; nb < 4; ++nb) {
      v16h bf = load_b_frag(pB, kb, nbBase + nb, NB, lane);
      acc[nb] = __builtin_amdgcn_wmma_f32_16x16x32_f16(
          false, a, false, bf, (short)0, acc[nb], false, false);
    }
  }
}

DEV void plucker6(const float a[4], const float b[4], float l[6]) {
  l[0] = a[0] * b[1] - a[1] * b[0];
  l[1] = a[0] * b[2] - a[2] * b[0];
  l[2] = a[0] * b[3] - a[3] * b[0];
  l[3] = a[1] * b[2] - a[2] * b[1];
  l[4] = a[1] * b[3] - a[3] * b[1];
  l[5] = a[2] * b[3] - a[3] * b[2];
  float s = l[0]*l[0] + l[1]*l[1] + l[2]*l[2] + l[3]*l[3] + l[4]*l[4] + l[5]*l[5];
  float inv = 1.f / fmaxf(sqrtf(s), 1e-12f);
#pragma unroll
  for (int i = 0; i < 6; ++i) l[i] *= inv;
}

#if PBA_HAVE_TDM
// TDM: DMA a 2D tile (64 rows x 32 f16, row stride T) from global into LDS.
// D# built per CDNA5 ISA 8.3/8.4: group0 = {flags, lds_addr, gaddr}, group1 =
// {mask/size, tensor_dim0/1, tile_dim0/1/2, strides}.  2D -> groups 2/3 zero.
DEV void tdm_load_vtile(const _Float16* gsrc, unsigned lds_off) {
  unsigned long long ga = (unsigned long long)(uintptr_t)gsrc;
  u32x4 g0;
  g0.x = 1u;                                              // count=1, user D#
  g0.y = lds_off;                                         // LDS byte address
  g0.z = (unsigned)(ga & 0xFFFFFFFFu);                    // global_addr[31:0]
  g0.w = (unsigned)((ga >> 32) & 0x01FFFFFFu) | (2u << 30); // [56:32] | type=2
  i32x8 g1;
  g1[0] = 1 << 16;                // workgroup_mask=0, data_size=1 (2B elems)
  g1[1] = (int)(2048u << 16);     // tensor_dim0 = T (contiguous key axis)
  g1[2] = (int)(64u << 16);       // tensor_dim0 hi=0 | tensor_dim1 = 64 rows
  g1[3] = (int)(32u << 16);       // tensor_dim1 hi=0 | tile_dim0 = 32 keys
  g1[4] = 64;                     // tile_dim1 = 64 rows, tile_dim2 = 0
  g1[5] = 2048;                   // tensor_dim0_stride = T
  g1[6] = 0;                      // stride hi | tensor_dim1_stride lo
  g1[7] = 0;
  i32x4 z4 = {0, 0, 0, 0};
#if defined(__clang_major__) && (__clang_major__ >= 23)
  i32x8 z8 = {0, 0, 0, 0, 0, 0, 0, 0};
  __builtin_amdgcn_tensor_load_to_lds(g0, g1, z4, z4, z8, 0);
#else
  __builtin_amdgcn_tensor_load_to_lds(g0, g1, z4, z4, 0);
#endif
}
#endif

// ---------------------------------------------------------------------------
// Prep kernels
// ---------------------------------------------------------------------------
__global__ void __launch_bounds__(256) cvt_f16_kernel(const float* __restrict__ in,
                                                      _Float16* __restrict__ out, int n) {
  int i = blockIdx.x * blockDim.x + threadIdx.x;
  if (i < n) out[i] = (_Float16)in[i];
}

__global__ void __launch_bounds__(256) shift_prev_kernel(const float* __restrict__ x,
                                                         _Float16* __restrict__ out, int n) {
  using namespace pba;
  int i = blockIdx.x * blockDim.x + threadIdx.x;
  if (i >= n) return;
  int t = (i / D) % T;
  out[i] = (t == 0) ? (_Float16)0.f : (_Float16)x[i - D];
}

// Pack W (N x K, row-major, leading dim ldw, column offset coff) into WMMA
// B-fragment layout: out[((kb*N/16+nb)*32+lane)*16+i] =
//   W[nb*16 + lane%16][coff + kb*32 + i + 16*(lane>=16)]
__global__ void __launch_bounds__(256) pack_b_kernel(const float* __restrict__ W,
                                                     _Float16* __restrict__ out,
                                                     int N, int K, int ldw, int coff) {
  int idx = blockIdx.x * blockDim.x + threadIdx.x;
  if (idx >= N * K) return;
  int i    = idx & 15;
  int lane = (idx >> 4) & 31;
  int frag = idx >> 9;
  int NB = N >> 4;
  int nb = frag % NB;
  int kb = frag / NB;
  int rowW = nb * 16 + (lane & 15);
  int colW = coff + kb * 32 + i + ((lane & 16) ? 16 : 0);
  out[idx] = (_Float16)W[(size_t)rowW * ldw + colW];
}

// ---------------------------------------------------------------------------
// q/k projections + Plucker lines.  One wave per 16-row tile.
// ---------------------------------------------------------------------------
__global__ void __launch_bounds__(32) qk_plucker_kernel(
    const _Float16* __restrict__ xh, const _Float16* __restrict__ xph,
    const _Float16* __restrict__ pWq1a, const _Float16* __restrict__ pWq1b,
    const _Float16* __restrict__ pWq2a, const _Float16* __restrict__ pWq2b,
    const _Float16* __restrict__ pWk1,  const _Float16* __restrict__ pWk2,
    float* __restrict__ q8, float* __restrict__ jk8) {
  using namespace pba;
  __shared__ float lds[4][16][64];
  const int lane = threadIdx.x;
  const int lg = lane >> 4, ln = lane & 15;
  const int tile = blockIdx.x;              // 0 .. TB/16-1
  const int b = tile / (T / 16);
  const int t0 = (tile % (T / 16)) * 16;
  const int row0 = tile * 16;               // global row over B*T

  v8f acc[4];
  // q1 = x @ W1q[:, :D]^T + x_prev @ W1q[:, D:]^T
  zero_acc4(acc);
  gemm_16x64(xh,  D, row0, pWq1a, 0, 4, 32, lane, acc);
  gemm_16x64(xph, D, row0, pWq1b, 0, 4, 32, lane, acc);
#pragma unroll
  for (int nb = 0; nb < 4; ++nb)
#pragma unroll
    for (int r = 0; r < 8; ++r) lds[0][r + 8 * lg][nb * 16 + ln] = acc[nb][r];
  // q2
  zero_acc4(acc);
  gemm_16x64(xh,  D, row0, pWq2a, 0, 4, 32, lane, acc);
  gemm_16x64(xph, D, row0, pWq2b, 0, 4, 32, lane, acc);
#pragma unroll
  for (int nb = 0; nb < 4; ++nb)
#pragma unroll
    for (int r = 0; r < 8; ++r) lds[1][r + 8 * lg][nb * 16 + ln] = acc[nb][r];
  // k1
  zero_acc4(acc);
  gemm_16x64(xh, D, row0, pWk1, 0, 4, 32, lane, acc);
#pragma unroll
  for (int nb = 0; nb < 4; ++nb)
#pragma unroll
    for (int r = 0; r < 8; ++r) lds[2][r + 8 * lg][nb * 16 + ln] = acc[nb][r];
  // k2
  zero_acc4(acc);
  gemm_16x64(xh, D, row0, pWk2, 0, 4, 32, lane, acc);
#pragma unroll
  for (int nb = 0; nb < 4; ++nb)
#pragma unroll
    for (int r = 0; r < 8; ++r) lds[3][r + 8 * lg][nb * 16 + ln] = acc[nb][r];

  __syncthreads();

  // 16 rows x 16 heads = 256 (t,h) pairs; 8 per lane.
  for (int it = 0; it < 8; ++it) {
    int p = it * 32 + lane;
    int tr = p >> 4;
    int hh = p & 15;
    float a[4], c[4], e[4], f[4];
#pragma unroll
    for (int j = 0; j < 4; ++j) {
      a[j] = lds[0][tr][hh * 4 + j];
      c[j] = lds[1][tr][hh * 4 + j];
      e[j] = lds[2][tr][hh * 4 + j];
      f[j] = lds[3][tr][hh * 4 + j];
    }
    float ql[6], kl[6];
    plucker6(a, c, ql);
    plucker6(e, f, kl);
    size_t base = (((size_t)b * H + hh) * T + (t0 + tr)) * 8;
    q8[base + 0] = ql[0]; q8[base + 1] = ql[1]; q8[base + 2] = ql[2];
    q8[base + 3] = ql[3]; q8[base + 4] = ql[4]; q8[base + 5] = ql[5];
    q8[base + 6] = 0.f;   q8[base + 7] = 0.f;
    // Jk = k @ J6^T : reversal with sign flips on channels 1 and 4
    jk8[base + 0] =  kl[5]; jk8[base + 1] = -kl[4]; jk8[base + 2] =  kl[3];
    jk8[base + 3] =  kl[2]; jk8[base + 4] = -kl[1]; jk8[base + 5] =  kl[0];
    jk8[base + 6] = 0.f;    jk8[base + 7] = 0.f;
  }
}

// ---------------------------------------------------------------------------
// v = x @ Wv^T + bv, stored transposed per head: vt[b][h][d][t] (f16)
// ---------------------------------------------------------------------------
__global__ void __launch_bounds__(32) v_proj_kernel(const _Float16* __restrict__ xh,
                                                    const _Float16* __restrict__ pWv,
                                                    const float* __restrict__ bv,
                                                    _Float16* __restrict__ vt) {
  using namespace pba;
  const int lane = threadIdx.x;
  const int lg = lane >> 4, ln = lane & 15;
  const int rt = blockIdx.x;                // row tile over B*T
  const int cg = blockIdx.y;                // 64-col group == head (D/64 == H)
  v8f acc[4];
  zero_acc4(acc);
  gemm_16x64(xh, D, rt * 16, pWv, cg * 4, D / 16, D / 32, lane, acc);
  const int b = rt / (T / 16);
  const int t0 = (rt % (T / 16)) * 16;
#pragma unroll
  for (int nb = 0; nb < 4; ++nb) {
    int dloc = nb * 16 + ln;                // d within head
    float bias = bv[cg * 64 + dloc];
    v8h w;
#pragma unroll
    for (int r = 0; r < 8; ++r) w[r] = (_Float16)(acc[nb][r] + bias);
    size_t off = (((size_t)b * H + cg) * 64 + dloc) * (size_t)T + t0 + 8 * lg;
    *(v8h*)(vt + off) = w;                  // consecutive t -> one b128 store
  }
}

// ---------------------------------------------------------------------------
// Flash attention, transposed tiles.  One wave handles 16 queries of one (b,h).
// V^T tiles are staged in LDS by the Tensor Data Mover, double buffered.
// ---------------------------------------------------------------------------
__global__ void __launch_bounds__(32) attn_kernel(const float* __restrict__ q8,
                                                  const float* __restrict__ jk8,
                                                  const _Float16* __restrict__ vt,
                                                  _Float16* __restrict__ oh) {
  using namespace pba;
  const int lane = threadIdx.x;
  const int lg = lane >> 4;
  const int ln = lane & 15;
  const int ntile = T / 16;
  const int t0 = (blockIdx.x % ntile) * 16;
  const int bh = blockIdx.x / ntile;
  const int h = bh % H, b = bh / H;
  const float* qb  = q8  + (size_t)bh * T * 8;
  const float* kjb = jk8 + (size_t)bh * T * 8;
  const _Float16* vb = vt + (size_t)bh * 64 * T;

  // B fragments of q^T (4x16 f32): lane holds query ln, K = i + 2*lg
  const v2f bq0 = *(const v2f*)(qb + (size_t)(t0 + ln) * 8 + 2 * lg);
  const v2f bq1 = *(const v2f*)(qb + (size_t)(t0 + ln) * 8 + 4 + 2 * lg);

  float m = -1e30f, lsum = 0.f;
  v8f o[4];
  zero_acc4(o);

  const int nblk = (t0 + 16 + 31) / 32;
#if PBA_HAVE_TDM
  __shared__ __attribute__((aligned(16))) _Float16 vstage[2][64 * 32];
  tdm_load_vtile(vb, (unsigned)(uintptr_t)&vstage[0][0]);   // prime buffer 0
#endif

  for (int iblk = 0; iblk < nblk; ++iblk) {
    const int s0 = iblk * 32;
#if PBA_HAVE_TDM
    if (iblk + 1 < nblk)   // kick off next tile's DMA before doing any math
      tdm_load_vtile(vb + (size_t)(s0 + 32),
                     (unsigned)(uintptr_t)&vstage[(iblk + 1) & 1][0]);
#endif
    if (iblk + 1 < nblk)
      __builtin_prefetch(kjb + (size_t)(s0 + 32 + ln) * 8, 0, 0);

    float l0[8], l1[8];   // S^T element (key = s0 [+16] + r + 8*lg, query = t0+ln)
#pragma unroll
    for (int tt = 0; tt < 2; ++tt) {
      const float* kr = kjb + (size_t)(s0 + tt * 16 + ln) * 8;
      v2f a0 = *(const v2f*)(kr + 2 * lg);       // Jk A-frag, channels 0-3
      v2f a1 = *(const v2f*)(kr + 4 + 2 * lg);   // channels 4-7 (6,7 zero pad)
      v8f z = {0.f, 0.f, 0.f, 0.f, 0.f, 0.f, 0.f, 0.f};
      z = __builtin_amdgcn_wmma_f32_16x16x4_f32(false, a1, false, bq1, (short)0, z, false, false);
      z = __builtin_amdgcn_wmma_f32_16x16x4_f32(false, a0, false, bq0, (short)0, z, false, false);
#pragma unroll
      for (int r = 0; r < 8; ++r) { if (tt == 0) l0[r] = z[r]; else l1[r] = z[r]; }
    }

    const int q = t0 + ln;
    float mb = -1e30f;
#pragma unroll
    for (int r = 0; r < 8; ++r) {
      int k0 = s0 + r + 8 * lg;
      int k1 = s0 + 16 + r + 8 * lg;
      l0[r] = (k0 <= q) ? l0[r] * SCALE : -1e30f;   // causal mask
      l1[r] = (k1 <= q) ? l1[r] * SCALE : -1e30f;
      mb = fmaxf(mb, fmaxf(l0[r], l1[r]));
    }
    mb = fmaxf(mb, __shfl_xor(mb, 16));             // combine lane halves
    float mn = fmaxf(m, mb);
    float corr = __expf(m - mn);
    float p0[8], p1[8], s = 0.f;
#pragma unroll
    for (int r = 0; r < 8; ++r) {
      p0[r] = __expf(l0[r] - mn);
      p1[r] = __expf(l1[r] - mn);
      s += p0[r] + p1[r];
    }
    s += __shfl_xor(s, 16);
    lsum = lsum * corr + s;
    m = mn;
#pragma unroll
    for (int dt = 0; dt < 4; ++dt) o[dt] = o[dt] * corr;

    // Build B fragment of P^T (32x16 f16): lane ln holds its own query's probs;
    // the other half-wave's 8 key values come via one shfl_xor(16) per r.
    v16h pb;
#pragma unroll
    for (int r = 0; r < 8; ++r) {
      float o0 = __shfl_xor(p0[r], 16);
      float o1 = __shfl_xor(p1[r], 16);
      pb[r]     = (_Float16)(lg ? o1 : p0[r]);      // K = r + 16*lg
      pb[r + 8] = (_Float16)(lg ? p1[r] : o0);      // K = 8 + r + 16*lg
    }

    // o^T += V^T(16d x 32k) @ P^T(32k x 16q), 4 d-tiles
#if PBA_HAVE_TDM
    if (iblk + 1 < nblk) __builtin_amdgcn_s_wait_tensorcnt(1);  // cur tile done
    else                 __builtin_amdgcn_s_wait_tensorcnt(0);
    const _Float16* vsrc = &vstage[iblk & 1][0];
#pragma unroll
    for (int dt = 0; dt < 4; ++dt) {
      v16h a = load_a_f16(vsrc, 32, dt * 16 + ln, 0, lane);     // ds_load_b128
      o[dt] = __builtin_amdgcn_wmma_f32_16x16x32_f16(
          false, a, false, pb, (short)0, o[dt], false, false);
    }
#else
#pragma unroll
    for (int dt = 0; dt < 4; ++dt) {
      v16h a = load_a_f16(vb, T, dt * 16 + ln, s0, lane);
      o[dt] = __builtin_amdgcn_wmma_f32_16x16x32_f16(
          false, a, false, pb, (short)0, o[dt], false, false);
    }
#endif
  }

  float rl = 1.f / lsum;                             // lane's query == ln
#pragma unroll
  for (int dt = 0; dt < 4; ++dt) {
    v8h w;
#pragma unroll
    for (int r = 0; r < 8; ++r) w[r] = (_Float16)(o[dt][r] * rl);
    size_t off = ((size_t)(b * T + t0 + ln)) * D + h * 64 + dt * 16 + 8 * lg;
    *(v8h*)(oh + off) = w;
  }
}

// ---------------------------------------------------------------------------
// out = oh @ Wo^T + bo (fp32 output)
// ---------------------------------------------------------------------------
__global__ void __launch_bounds__(32) out_proj_kernel(const _Float16* __restrict__ oh,
                                                      const _Float16* __restrict__ pWo,
                                                      const float* __restrict__ bo,
                                                      float* __restrict__ out) {
  using namespace pba;
  const int lane = threadIdx.x;
  const int lg = lane >> 4, ln = lane & 15;
  const int rt = blockIdx.x;                // row tile over B*T
  const int cg = blockIdx.y;                // 64-col group
  v8f acc[4];
  zero_acc4(acc);
  gemm_16x64(oh, D, rt * 16, pWo, cg * 4, D / 16, D / 32, lane, acc);
#pragma unroll
  for (int nb = 0; nb < 4; ++nb) {
    int col = cg * 64 + nb * 16 + ln;
    float bias = bo[col];
#pragma unroll
    for (int r = 0; r < 8; ++r) {
      size_t row = (size_t)rt * 16 + r + 8 * lg;
      out[row * D + col] = acc[nb][r] + bias;
    }
  }
}

// ---------------------------------------------------------------------------
// Host launcher
// ---------------------------------------------------------------------------
extern "C" void kernel_launch(void* const* d_in, const int* in_sizes, int n_in,
                              void* d_out, int out_size, void* d_ws, size_t ws_size,
                              hipStream_t stream) {
  using namespace pba;
  (void)in_sizes; (void)n_in; (void)out_size; (void)ws_size;
  const float* x   = (const float*)d_in[0];
  const float* W1q = (const float*)d_in[1];
  const float* W2q = (const float*)d_in[2];
  const float* W1k = (const float*)d_in[3];
  const float* W2k = (const float*)d_in[4];
  const float* Wv  = (const float*)d_in[5];
  const float* bv  = (const float*)d_in[6];
  const float* Wo  = (const float*)d_in[7];
  const float* bo  = (const float*)d_in[8];
  float* out = (float*)d_out;

  char* ws = (char*)d_ws;
  size_t off = 0;
  auto alloc = [&](size_t bytes) -> void* {
    void* p = ws + off;
    off = (off + bytes + 255) & ~(size_t)255;
    return p;
  };
  _Float16* xh   = (_Float16*)alloc((size_t)TB * D * 2);
  _Float16* xph  = (_Float16*)alloc((size_t)TB * D * 2);
  float*    q8   = (float*)   alloc((size_t)B * H * T * 8 * 4);
  float*    jk8  = (float*)   alloc((size_t)B * H * T * 8 * 4);
  _Float16* vt   = (_Float16*)alloc((size_t)B * H * 64 * T * 2);
  _Float16* oh   = (_Float16*)alloc((size_t)TB * D * 2);
  _Float16* pWq1a = (_Float16*)alloc((size_t)64 * 1024 * 2);
  _Float16* pWq1b = (_Float16*)alloc((size_t)64 * 1024 * 2);
  _Float16* pWq2a = (_Float16*)alloc((size_t)64 * 1024 * 2);
  _Float16* pWq2b = (_Float16*)alloc((size_t)64 * 1024 * 2);
  _Float16* pWk1  = (_Float16*)alloc((size_t)64 * 1024 * 2);
  _Float16* pWk2  = (_Float16*)alloc((size_t)64 * 1024 * 2);
  _Float16* pWv   = (_Float16*)alloc((size_t)1024 * 1024 * 2);
  _Float16* pWo   = (_Float16*)alloc((size_t)1024 * 1024 * 2);

  const int n = TB * D;
  cvt_f16_kernel<<<(n + 255) / 256, 256, 0, stream>>>(x, xh, n);
  shift_prev_kernel<<<(n + 255) / 256, 256, 0, stream>>>(x, xph, n);

  auto pack = [&](const float* W, _Float16* dst, int N, int K, int ldw, int coff) {
    int tot = N * K;
    pack_b_kernel<<<(tot + 255) / 256, 256, 0, stream>>>(W, dst, N, K, ldw, coff);
  };
  pack(W1q, pWq1a, 64, 1024, 2048, 0);
  pack(W1q, pWq1b, 64, 1024, 2048, 1024);
  pack(W2q, pWq2a, 64, 1024, 2048, 0);
  pack(W2q, pWq2b, 64, 1024, 2048, 1024);
  pack(W1k, pWk1, 64, 1024, 1024, 0);
  pack(W2k, pWk2, 64, 1024, 1024, 0);
  pack(Wv,  pWv, 1024, 1024, 1024, 0);
  pack(Wo,  pWo, 1024, 1024, 1024, 0);

  qk_plucker_kernel<<<TB / 16, 32, 0, stream>>>(xh, xph, pWq1a, pWq1b, pWq2a,
                                                pWq2b, pWk1, pWk2, q8, jk8);
  v_proj_kernel<<<dim3(TB / 16, D / 64), 32, 0, stream>>>(xh, pWv, bv, vt);
  attn_kernel<<<B * H * (T / 16), 32, 0, stream>>>(q8, jk8, vt, oh);
  out_proj_kernel<<<dim3(TB / 16, D / 64), 32, 0, stream>>>(oh, pWo, bo, out);
}
